// SelfSeqAtten_16166256902633
// MI455X (gfx1250) — matrix operand order repaired and checked by reading, not compiled, not run on hardware
//
#include <hip/hip_runtime.h>
#include <hip/hip_bf16.h>
#include <stdint.h>

#define C_LEN 2048
#define BATCH 8
#define DDIM  128
#define ROWS_PER_WAVE 16
#define NWAVES 4
#define ROWS_PER_WG (ROWS_PER_WAVE * NWAVES)   // 64
#define JTILE 32
#define NJT (C_LEN / JTILE)                    // 64
#define QSTR 136                                // padded LDS row stride (elems)
#define KSTR 136
#define VSTR 40
#define L2E 1.4426950408889634f

typedef __attribute__((ext_vector_type(16))) __bf16 v16bf;
typedef __attribute__((ext_vector_type(8)))  float  v8f;

union Frag { v16bf v; unsigned int u[8]; };

__device__ __forceinline__ unsigned short f2bf(float f) {
    unsigned int u = __float_as_uint(f);
    u += 0x7FFFu + ((u >> 16) & 1u);           // round-to-nearest-even
    return (unsigned short)(u >> 16);
}
__device__ __forceinline__ unsigned int pk2bf(float a, float b) {
    return (unsigned int)f2bf(a) | ((unsigned int)f2bf(b) << 16);
}

__device__ __forceinline__ v8f wmma_bf16(v16bf a, v16bf b, v8f c) {
    return __builtin_amdgcn_wmma_f32_16x16x32_bf16(false, a, false, b,
                                                   (short)0, c, false, false);
}

__device__ __forceinline__ void load_frag16(Frag& f, const unsigned short* p0,
                                            const unsigned short* p1) {
    uint4 a = *(const uint4*)p0;
    uint4 c = *(const uint4*)p1;
    f.u[0] = a.x; f.u[1] = a.y; f.u[2] = a.z; f.u[3] = a.w;
    f.u[4] = c.x; f.u[5] = c.y; f.u[6] = c.z; f.u[7] = c.w;
}

// ---------------- pre-pass: f32 -> bf16, build Kb[b][c][d] and Vt[b][d][c]
__global__ void __launch_bounds__(256)
cvt_kernel(const float* __restrict__ h,
           unsigned short* __restrict__ Kb,
           unsigned short* __restrict__ Vt) {
    int f4 = blockIdx.x * 256 + threadIdx.x;        // 0 .. 524287
    int e  = f4 * 4;                                // element index into h
    int d  = e & (DDIM - 1);
    int b  = (e >> 7) & (BATCH - 1);
    int c  = e >> 10;
    float4 v = *(const float4*)(h + e);
    unsigned short s0 = f2bf(v.x), s1 = f2bf(v.y), s2 = f2bf(v.z), s3 = f2bf(v.w);
    // K: row-major per batch
    unsigned short* kp = Kb + ((size_t)(b * C_LEN + c) * DDIM + d);
    uint2 kw; kw.x = (unsigned)s0 | ((unsigned)s1 << 16);
    kw.y = (unsigned)s2 | ((unsigned)s3 << 16);
    *(uint2*)kp = kw;
    // V^T: feature-major per batch (scatter)
    size_t vbase = (size_t)(b * DDIM + d) * C_LEN + c;
    Vt[vbase]              = s0;
    Vt[vbase + C_LEN]      = s1;
    Vt[vbase + 2 * C_LEN]  = s2;
    Vt[vbase + 3 * C_LEN]  = s3;
}

// ---------------- fused flash attention
__global__ void __launch_bounds__(128)
attn_kernel(const float* __restrict__ h, const int* __restrict__ hmask,
            const unsigned short* __restrict__ Kb,
            const unsigned short* __restrict__ Vt,
            float* __restrict__ out) {
    __shared__ unsigned short Qs[ROWS_PER_WG * QSTR];
    __shared__ unsigned short Ks[2][JTILE * KSTR];
    __shared__ unsigned short Vs[2][DDIM * VSTR];
    __shared__ unsigned int   mbits[2];

    const int tid  = threadIdx.x;
    const int w    = tid >> 5;
    const int lane = tid & 31;
    const int b    = blockIdx.y;
    const int wg_row0 = blockIdx.x * ROWS_PER_WG;
    const int row0    = wg_row0 + w * ROWS_PER_WAVE;

    const int qloc = lane & 15;     // query (N) index for this lane
    const int hg   = lane >> 4;     // lane-half
    const int fb   = hg * 8;        // A-operand K-offset base

    const unsigned ksOff0 = (unsigned)(uintptr_t)(&Ks[0][0]);
    const unsigned vsOff0 = (unsigned)(uintptr_t)(&Vs[0][0]);

    // ---- stage this WG's 64 Q rows into LDS as bf16 (row-major, padded)
    #pragma unroll
    for (int it = 0; it < 16; ++it) {
        int idx = it * 128 + tid;           // 2048 float4s
        int r   = idx >> 5;                 // local row 0..63
        int c4  = idx & 31;                 // float4 within row
        int grow = wg_row0 + r;
        const float4 vq = *(const float4*)(h + ((size_t)(grow * BATCH + b) * DDIM + c4 * 4));
        uint2 qw; qw.x = pk2bf(vq.x, vq.y); qw.y = pk2bf(vq.z, vq.w);
        *(uint2*)(&Qs[r * QSTR + c4 * 4]) = qw;
    }

    // ---- async-stage one 32-key tile (K rows + V^T rows) into LDS buffer
    auto issue_tile = [&](int jt, int buf) {
        const int j0 = jt * JTILE;
        const unsigned kl = ksOff0 + (unsigned)buf * (JTILE * KSTR * 2);
        #pragma unroll
        for (int i = 0; i < 4; ++i) {                  // 8 K rows per wave
            int chunk = i * 32 + lane;                 // 128 x 16B chunks
            int rl    = chunk >> 4;                    // 0..7
            int c16   = chunk & 15;
            int grow  = j0 + w * 8 + rl;
            unsigned goff = (unsigned)((b * C_LEN + grow) * (DDIM * 2) + c16 * 16);
            unsigned loff = kl + (unsigned)((w * 8 + rl) * (KSTR * 2) + c16 * 16);
            asm volatile("global_load_async_to_lds_b128 %0, %1, %2"
                         :: "v"(loff), "v"(goff), "s"(Kb) : "memory");
        }
        const unsigned vl = vsOff0 + (unsigned)buf * (DDIM * VSTR * 2);
        #pragma unroll
        for (int i = 0; i < 4; ++i) {                  // 32 V^T rows per wave
            int chunk = i * 32 + lane;
            int fl    = chunk >> 2;                    // 0..31
            int c4    = chunk & 3;
            int frow  = w * 32 + fl;
            unsigned goff = (unsigned)((b * DDIM + frow) * (C_LEN * 2) + j0 * 2 + c4 * 16);
            unsigned loff = vl + (unsigned)(frow * (VSTR * 2) + c4 * 16);
            asm volatile("global_load_async_to_lds_b128 %0, %1, %2"
                         :: "v"(loff), "v"(goff), "s"(Vt) : "memory");
        }
        if (w == 0) {   // key-valid bitmask for this tile
            unsigned long long bal = __ballot(hmask[b * C_LEN + j0 + lane] != 0);
            if (lane == 0) mbits[buf] = (unsigned)bal;
        }
    };

    issue_tile(0, 0);
    __syncthreads();   // Q staging visible (async tile-0 still in flight)

    // ---- build loop-invariant Q fragments (B-operand layout:
    //      lane-half hg holds K = hg*8..hg*8+7 and 16+hg*8..16+hg*8+7 of each 32-chunk)
    Frag Qf[4];
    {
        const unsigned short* qp = &Qs[(w * 16 + qloc) * QSTR];
        #pragma unroll
        for (int c = 0; c < 4; ++c) {
            const unsigned short* p = qp + c * 32 + hg * 8;
            load_frag16(Qf[c], p, p + 16);
        }
    }

    v8f acc[8];
    #pragma unroll
    for (int i = 0; i < 8; ++i)
        #pragma unroll
        for (int e = 0; e < 8; ++e) acc[i][e] = 0.0f;
    float mrun = -1e30f, lrun = 0.0f;
    const int qglob = row0 + qloc;

    for (int jt = 0; jt < NJT; ++jt) {
        const int buf = jt & 1;
        if (jt + 1 < NJT) {
            issue_tile(jt + 1, 1 - buf);
            asm volatile("s_wait_asynccnt 8" ::: "memory");   // current tile done
        } else {
            asm volatile("s_wait_asynccnt 0" ::: "memory");
        }
        __syncthreads();
        const unsigned vm = mbits[buf];
        const int j0 = jt * JTILE;

        // ---- S^T = K_tile · Q^T  (two 16-key halves, K=128 over 4 chunks)
        v8f S0, S1;
        #pragma unroll
        for (int e = 0; e < 8; ++e) { S0[e] = 0.0f; S1[e] = 0.0f; }
        const unsigned short* kbp = &Ks[buf][0];
        #pragma unroll
        for (int c = 0; c < 4; ++c) {
            Frag A0, A1;
            const unsigned short* pa = kbp + qloc * KSTR + c * 32 + fb;
            const unsigned short* pb = kbp + (16 + qloc) * KSTR + c * 32 + fb;
            load_frag16(A0, pa, pa + 16);
            load_frag16(A1, pb, pb + 16);
            S0 = wmma_bf16(A0.v, Qf[c].v, S0);
            S1 = wmma_bf16(A1.v, Qf[c].v, S1);
        }

        // ---- diagonal zero, key mask, online softmax (lane pair q / q+16)
        float p[16];
        float mt = -1e30f;
        #pragma unroll
        for (int r = 0; r < 8; ++r) {
            float s0 = S0[r], s1 = S1[r];
            int k0 = j0 + fb + r, k1 = k0 + 16;
            if (k0 == qglob) s0 = 0.0f;
            if (k1 == qglob) s1 = 0.0f;
            s0 = ((vm >> (fb + r)) & 1u)      ? s0 : -1e30f;
            s1 = ((vm >> (16 + fb + r)) & 1u) ? s1 : -1e30f;
            p[r] = s0; p[8 + r] = s1;
            mt = fmaxf(mt, fmaxf(s0, s1));
        }
        mt = fmaxf(mt, __shfl_xor(mt, 16, 32));
        const float mnew  = fmaxf(mrun, mt);
        const float scale = exp2f((mrun - mnew) * L2E);

        float psum = 0.0f;
        float ex[16];
        #pragma unroll
        for (int i = 0; i < 16; ++i) {
            float e = exp2f((p[i] - mnew) * L2E);
            e = (p[i] > -1e29f) ? e : 0.0f;
            ex[i] = e;
            psum += e;
        }
        psum += __shfl_xor(psum, 16, 32);
        lrun = lrun * scale + psum;
        mrun = mnew;
        #pragma unroll
        for (int i = 0; i < 8; ++i)
            #pragma unroll
            for (int e = 0; e < 8; ++e) acc[i][e] *= scale;

        // ---- pack P^T into B-operand layout (own keys, no lane exchange)
        unsigned own_lo[4], own_hi[4];
        #pragma unroll
        for (int i = 0; i < 4; ++i) {
            own_lo[i] = pk2bf(ex[2 * i],     ex[2 * i + 1]);      // T0 keys
            own_hi[i] = pk2bf(ex[8 + 2 * i], ex[9 + 2 * i]);      // T1 keys
        }
        Frag P;
        #pragma unroll
        for (int i = 0; i < 4; ++i) {
            P.u[i]     = own_lo[i];
            P.u[4 + i] = own_hi[i];
        }

        // ---- O^T += V^T_chunk · P^T
        const unsigned short* vbp = &Vs[buf][0];
        #pragma unroll
        for (int dc = 0; dc < 8; ++dc) {
            Frag A;
            const unsigned short* pv = vbp + (dc * 16 + qloc) * VSTR + fb;
            load_frag16(A, pv, pv + 16);
            acc[dc] = wmma_bf16(A.v, P.v, acc[dc]);
        }
        __syncthreads();   // buffer reuse fence before next issue
    }

    // ---- normalize and store: lane holds O[qglob][dc*16 + hg*8 + 0..7]
    const float inv = 1.0f / lrun;
    float* orow = out + ((size_t)(qglob * BATCH + b) * DDIM);
    #pragma unroll
    for (int dc = 0; dc < 8; ++dc) {
        float* po = orow + dc * 16 + hg * 8;
        float4 o0, o1;
        o0.x = acc[dc][0] * inv; o0.y = acc[dc][1] * inv;
        o0.z = acc[dc][2] * inv; o0.w = acc[dc][3] * inv;
        o1.x = acc[dc][4] * inv; o1.y = acc[dc][5] * inv;
        o1.z = acc[dc][6] * inv; o1.w = acc[dc][7] * inv;
        *(float4*)(po)     = o0;
        *(float4*)(po + 4) = o1;
    }
}

extern "C" void kernel_launch(void* const* d_in, const int* in_sizes, int n_in,
                              void* d_out, int out_size, void* d_ws, size_t ws_size,
                              hipStream_t stream) {
    const float* h  = (const float*)d_in[0];
    const int*   hm = (const int*)d_in[1];
    unsigned short* Kb = (unsigned short*)d_ws;                                  // 4 MB
    unsigned short* Vt = (unsigned short*)((char*)d_ws +
                                           (size_t)BATCH * C_LEN * DDIM * 2);    // 4 MB
    cvt_kernel<<<dim3((C_LEN * BATCH * DDIM / 4) / 256), 256, 0, stream>>>(h, Kb, Vt);
    dim3 grid(C_LEN / ROWS_PER_WG, BATCH);
    attn_kernel<<<grid, 128, 0, stream>>>(h, hm, Kb, Vt, (float*)d_out);
}
